// GATLayer_4741643895566
// MI455X (gfx1250) — matrix operand order repaired and checked by reading, not compiled, not run on hardware
//
#include <hip/hip_runtime.h>
#include <cstdint>

// GAT layer for MI455X (gfx1250, wave32, WMMA, async-to-LDS).
// N=8192, IN_F=512, OUT_F=128.
// adj+M = 512 MB -> single HBM pass, flash-style online softmax fused with
// v_wmma_f32_16x16x32_f16. Column dim split 4-ways (2048 waves) for MLP;
// adj/M streamed through a 4-deep LDS ring via global_load_async_to_lds_b128
// (ASYNCcnt-counted waits), so each wave keeps 32 async b128s in flight.

#define N_NODES 8192
#define IN_FEAT 512
#define OUT_FEAT 128
#define NGRP 4                   // column split
#define COLS_PER_GRP (N_NODES / NGRP)   // 2048
#define NTILE (COLS_PER_GRP / 32)       // 64 tiles of 32 columns

typedef __attribute__((ext_vector_type(16))) _Float16 v16h;
typedef __attribute__((ext_vector_type(8)))  _Float16 v8h;
typedef __attribute__((ext_vector_type(8)))  float    v8f;
typedef __attribute__((ext_vector_type(4)))  float    v4f;

// workspace layout (bytes)
#define WT_OFF   0                                         // f16 Wt[128][512]
#define HT_OFF   (WT_OFF + OUT_FEAT * IN_FEAT * 2)         // f16 Ht[128][8192]
#define S_OFF    (HT_OFF + OUT_FEAT * N_NODES * 2)         // f32 s[8192]
#define T_OFF    (S_OFF + N_NODES * 4)                     // f32 t[8192]
#define PACC_OFF (T_OFF + N_NODES * 4)                     // f32 pacc[512*4][16][128]
#define PST_OFF  (PACC_OFF + 512 * NGRP * 16 * OUT_FEAT * 4) // f32 pstat[512*4][16][2]
// total ~18.6 MB

#define WAIT_ASYNC(n) asm volatile("s_wait_asynccnt " #n ::: "memory")

// ---------------------------------------------------------------------------
// Kernel 0: Wt[n][k] = (f16) W[k][n]
// ---------------------------------------------------------------------------
__global__ void k_wtrans(const float* __restrict__ W, _Float16* __restrict__ Wt) {
    int idx = blockIdx.x * blockDim.x + threadIdx.x;
    int n = idx / IN_FEAT;
    int k = idx % IN_FEAT;
    Wt[idx] = (_Float16)W[k * OUT_FEAT + n];
}

// ---------------------------------------------------------------------------
// Kernel 1: h = X @ W via WMMA; emit Ht (f16, transposed) and s,t dot products.
// ---------------------------------------------------------------------------
__global__ void k_hmm(const float* __restrict__ X,
                      const _Float16* __restrict__ Wt,
                      const float* __restrict__ a_self,
                      const float* __restrict__ a_nei,
                      _Float16* __restrict__ Ht,
                      float* __restrict__ s_arr,
                      float* __restrict__ t_arr) {
    const int lane = threadIdx.x & 31;
    const int wave = threadIdx.x >> 5;
    const int rb   = blockIdx.x * 8 + wave;     // 0..511
    const int i0   = rb * 16;
    const int m    = lane & 15;
    const int hi   = lane >> 4;

    v8f acc[8];
#pragma unroll
    for (int t = 0; t < 8; ++t)
#pragma unroll
        for (int r = 0; r < 8; ++r) acc[t][r] = 0.0f;

    const float* xrow = X + (size_t)(i0 + m) * IN_FEAT;

    for (int k0 = 0; k0 < IN_FEAT; k0 += 32) {
        v4f a0 = *(const v4f*)(xrow + k0 + 16 * hi);
        v4f a1 = *(const v4f*)(xrow + k0 + 16 * hi + 4);
        v4f a2 = *(const v4f*)(xrow + k0 + 16 * hi + 8);
        v4f a3 = *(const v4f*)(xrow + k0 + 16 * hi + 12);
        v16h A;
#pragma unroll
        for (int j = 0; j < 4; ++j) {
            A[j]      = (_Float16)a0[j];
            A[4 + j]  = (_Float16)a1[j];
            A[8 + j]  = (_Float16)a2[j];
            A[12 + j] = (_Float16)a3[j];
        }
#pragma unroll
        for (int t = 0; t < 8; ++t) {
            const v16h B = *(const v16h*)(Wt + (size_t)(t * 16 + m) * IN_FEAT + k0 + 16 * hi);
            acc[t] = __builtin_amdgcn_wmma_f32_16x16x32_f16(
                false, A, false, B, (short)0, acc[t], false, false);
        }
    }

    float ps[8], pt[8];
#pragma unroll
    for (int r = 0; r < 8; ++r) { ps[r] = 0.0f; pt[r] = 0.0f; }
#pragma unroll
    for (int t = 0; t < 8; ++t) {
        const int n = t * 16 + m;
        const float as = a_self[n];
        const float an = a_nei[n];
#pragma unroll
        for (int r = 0; r < 8; ++r) {
            const float v = acc[t][r];
            Ht[(size_t)n * N_NODES + i0 + r + 8 * hi] = (_Float16)v;
            ps[r] += v * as;
            pt[r] += v * an;
        }
    }
#pragma unroll
    for (int r = 0; r < 8; ++r) {
#pragma unroll
        for (int off = 1; off < 16; off <<= 1) {
            ps[r] += __shfl_xor(ps[r], off, 32);
            pt[r] += __shfl_xor(pt[r], off, 32);
        }
    }
    if (m == 0) {
#pragma unroll
        for (int r = 0; r < 8; ++r) {
            s_arr[i0 + r + 8 * hi] = ps[r];
            t_arr[i0 + r + 8 * hi] = pt[r];
        }
    }
}

// ---------------------------------------------------------------------------
// Kernel 2: partial flash attention over one column group.
// blockIdx: g = blockIdx & 3 (column group), 8 waves = 8 consecutive row blocks
// sharing the same Ht slice (L1/L2 broadcast). adj/M stream through a 4-deep
// per-wave LDS ring filled with global_load_async_to_lds_b128.
// LDS: 8 waves * 4 bufs * (2KB adj + 2KB M) = 128 KB dynamic.
// ---------------------------------------------------------------------------
__global__ void k_attn(const float* __restrict__ adj,
                       const float* __restrict__ Mm,
                       const _Float16* __restrict__ Ht,
                       const float* __restrict__ s_arr,
                       const float* __restrict__ t_arr,
                       float* __restrict__ pacc,
                       float* __restrict__ pstat) {
    extern __shared__ char smem[];
    const int lane = threadIdx.x & 31;
    const int wave = threadIdx.x >> 5;
    const int g    = blockIdx.x & 3;                 // column group
    const int rb   = (blockIdx.x >> 2) * 8 + wave;   // 0..511
    const int i0   = rb * 16;
    const int m    = lane & 15;
    const int hi   = lane >> 4;
    const int jbase = g * COLS_PER_GRP;

    const float s_m = s_arr[i0 + m];

    float rowmax = -3.0e38f;
    float rowsum = 0.0f;
    v8f acc[8];
#pragma unroll
    for (int t = 0; t < 8; ++t)
#pragma unroll
        for (int r = 0; r < 8; ++r) acc[t][r] = 0.0f;

    // LDS ring addressing (AS3 offset = low 32 bits of the generic pointer)
    const unsigned ldsWave = (unsigned)(uintptr_t)smem + (unsigned)wave * 16384u;
    const unsigned ldsLane = ldsWave + (unsigned)lane * 64u;
    // 32-bit byte offset of this lane's row segment within adj/M
    const unsigned rowByte = (unsigned)(((i0 + m) * N_NODES + jbase + 8 * hi) * 4);
    const unsigned long long adjB = (unsigned long long)(uintptr_t)adj;
    const unsigned long long mmB  = (unsigned long long)(uintptr_t)Mm;

    auto issueTile = [&](int it) {
        const unsigned la = ldsLane + (unsigned)(it & 3) * 4096u;   // adj region
        const unsigned lm = la + 2048u;                             // M region
        const unsigned vo = rowByte + (unsigned)it * 128u;          // 32 floats / tile
        // slots 0-7 at +0,+16 ; slots 8-15 at +64,+80 (bytes)
        asm volatile("global_load_async_to_lds_b128 %0, %1, %2" :: "v"(la),       "v"(vo),      "s"(adjB) : "memory");
        asm volatile("global_load_async_to_lds_b128 %0, %1, %2" :: "v"(la + 16u), "v"(vo + 16u), "s"(adjB) : "memory");
        asm volatile("global_load_async_to_lds_b128 %0, %1, %2" :: "v"(la + 32u), "v"(vo + 64u), "s"(adjB) : "memory");
        asm volatile("global_load_async_to_lds_b128 %0, %1, %2" :: "v"(la + 48u), "v"(vo + 80u), "s"(adjB) : "memory");
        asm volatile("global_load_async_to_lds_b128 %0, %1, %2" :: "v"(lm),       "v"(vo),      "s"(mmB) : "memory");
        asm volatile("global_load_async_to_lds_b128 %0, %1, %2" :: "v"(lm + 16u), "v"(vo + 16u), "s"(mmB) : "memory");
        asm volatile("global_load_async_to_lds_b128 %0, %1, %2" :: "v"(lm + 32u), "v"(vo + 64u), "s"(mmB) : "memory");
        asm volatile("global_load_async_to_lds_b128 %0, %1, %2" :: "v"(lm + 48u), "v"(vo + 80u), "s"(mmB) : "memory");
    };

    auto processTile = [&](int it) {
        // B fragments first: 16 b128 loads in one batch (Ht slice is L1/L2-hot)
        v16h Bf[8];
#pragma unroll
        for (int t = 0; t < 8; ++t) {
            const _Float16* hp_ = Ht + (size_t)(t * 16 + m) * N_NODES + jbase + it * 32 + 8 * hi;
            Bf[t] = __builtin_shufflevector(*(const v8h*)(hp_), *(const v8h*)(hp_ + 16), 0, 1, 2, 3, 4, 5, 6, 7, 8, 9, 10, 11, 12, 13, 14, 15);
        }
        // t-vector (32 KB array, cache-resident)
        const float* tp = t_arr + jbase + it * 32 + 8 * hi;
        v4f tg0 = *(const v4f*)(tp);
        v4f tg1 = *(const v4f*)(tp + 4);
        v4f tg2 = *(const v4f*)(tp + 16);
        v4f tg3 = *(const v4f*)(tp + 20);
        // adj/M tiles from the LDS ring (this lane's own 64B region each)
        const float* tA = (const float*)(smem + (size_t)wave * 16384 + (size_t)(it & 3) * 4096) + lane * 16;
        const float* tM = tA + 512;     // +2048 bytes
        v4f ad0 = *(const v4f*)(tA);
        v4f ad1 = *(const v4f*)(tA + 4);
        v4f ad2 = *(const v4f*)(tA + 8);
        v4f ad3 = *(const v4f*)(tA + 12);
        v4f mg0 = *(const v4f*)(tM);
        v4f mg1 = *(const v4f*)(tM + 4);
        v4f mg2 = *(const v4f*)(tM + 8);
        v4f mg3 = *(const v4f*)(tM + 12);

        float e[16], msk[16];
#pragma unroll
        for (int j = 0; j < 4; ++j) {
            float er;
            er = (s_m + tg0[j]) * mg0[j]; er = er > 0.0f ? er : 0.2f * er;
            msk[j]      = ad0[j] > 0.0f ? 1.0f : 0.0f; e[j]      = ad0[j] > 0.0f ? er : -3.0e38f;
            er = (s_m + tg1[j]) * mg1[j]; er = er > 0.0f ? er : 0.2f * er;
            msk[4 + j]  = ad1[j] > 0.0f ? 1.0f : 0.0f; e[4 + j]  = ad1[j] > 0.0f ? er : -3.0e38f;
            er = (s_m + tg2[j]) * mg2[j]; er = er > 0.0f ? er : 0.2f * er;
            msk[8 + j]  = ad2[j] > 0.0f ? 1.0f : 0.0f; e[8 + j]  = ad2[j] > 0.0f ? er : -3.0e38f;
            er = (s_m + tg3[j]) * mg3[j]; er = er > 0.0f ? er : 0.2f * er;
            msk[12 + j] = ad3[j] > 0.0f ? 1.0f : 0.0f; e[12 + j] = ad3[j] > 0.0f ? er : -3.0e38f;
        }

        float tmax = e[0];
#pragma unroll
        for (int s = 1; s < 16; ++s) tmax = fmaxf(tmax, e[s]);
        tmax = fmaxf(tmax, __shfl_xor(tmax, 16, 32));
        const float mnew  = fmaxf(rowmax, tmax);
        const float alpha = __expf(rowmax - mnew);

        float psum = 0.0f;
        v16h P;
#pragma unroll
        for (int s = 0; s < 16; ++s) {
            const float p = msk[s] * __expf(e[s] - mnew);
            psum += p;
            P[s] = (_Float16)p;
        }
        psum += __shfl_xor(psum, 16, 32);
        rowsum = rowsum * alpha + psum;
        rowmax = mnew;

#pragma unroll
        for (int r = 0; r < 8; ++r) {
            const float fr = __shfl(alpha, r + 8 * hi, 32);
#pragma unroll
            for (int t = 0; t < 8; ++t) acc[t][r] *= fr;
        }
#pragma unroll
        for (int t = 0; t < 8; ++t)
            acc[t] = __builtin_amdgcn_wmma_f32_16x16x32_f16(
                false, P, false, Bf[t], (short)0, acc[t], false, false);
    };

    // prologue: fill the 4-deep ring (32 async b128s in flight)
    issueTile(0); issueTile(1); issueTile(2); issueTile(3);

    for (int it = 0; it < NTILE - 4; ++it) {
        WAIT_ASYNC(24);          // oldest tile's 8 transfers complete
        processTile(it);
        issueTile(it + 4);
    }
    WAIT_ASYNC(24); processTile(NTILE - 4);
    WAIT_ASYNC(16); processTile(NTILE - 3);
    WAIT_ASYNC(8);  processTile(NTILE - 2);
    WAIT_ASYNC(0);  processTile(NTILE - 1);

    // emit partial (unnormalized) accumulator + per-row stats
    const int pbase = (rb * NGRP + g) * 16;
#pragma unroll
    for (int r = 0; r < 8; ++r) {
        float* prow = pacc + (size_t)(pbase + r + 8 * hi) * OUT_FEAT + m;
#pragma unroll
        for (int t = 0; t < 8; ++t) prow[t * 16] = acc[t][r];
    }
    if (hi == 0) {
        pstat[(pbase + m) * 2 + 0] = rowmax;
        pstat[(pbase + m) * 2 + 1] = rowsum;
    }
}

// ---------------------------------------------------------------------------
// Kernel 3: merge the 4 column-group partials (log-sum-exp merge) + ELU.
// ---------------------------------------------------------------------------
__global__ void k_merge(const float* __restrict__ pacc,
                        const float* __restrict__ pstat,
                        float* __restrict__ out) {
    const int idx = blockIdx.x * blockDim.x + threadIdx.x;  // 0 .. 8192*128-1
    const int i = idx >> 7;
    const int n = idx & 127;
    const int rb = i >> 4;
    const int mrow = i & 15;

    float mg[NGRP], lg[NGRP];
    float M = -3.0e38f;
#pragma unroll
    for (int g = 0; g < NGRP; ++g) {
        const int pb = (rb * NGRP + g) * 16 + mrow;
        mg[g] = pstat[pb * 2 + 0];
        lg[g] = pstat[pb * 2 + 1];
        M = fmaxf(M, mg[g]);
    }
    float L = 0.0f, V = 0.0f;
#pragma unroll
    for (int g = 0; g < NGRP; ++g) {
        const float w = __expf(mg[g] - M);
        L += w * lg[g];
        V += w * pacc[(size_t)((rb * NGRP + g) * 16 + mrow) * OUT_FEAT + n];
    }
    const float hp = V / L;
    out[idx] = hp > 0.0f ? hp : (__expf(hp) - 1.0f);
}

// ---------------------------------------------------------------------------
extern "C" void kernel_launch(void* const* d_in, const int* in_sizes, int n_in,
                              void* d_out, int out_size, void* d_ws, size_t ws_size,
                              hipStream_t stream) {
    const float* X      = (const float*)d_in[0];   // [8192, 512]
    const float* adj    = (const float*)d_in[1];   // [8192, 8192]
    const float* Mm     = (const float*)d_in[2];   // [8192, 8192]
    const float* W      = (const float*)d_in[3];   // [512, 128]
    const float* a_self = (const float*)d_in[4];   // [128, 1]
    const float* a_nei  = (const float*)d_in[5];   // [128, 1]
    float* out = (float*)d_out;                    // [8192, 128]

    char* ws = (char*)d_ws;
    _Float16* Wt   = (_Float16*)(ws + WT_OFF);
    _Float16* Ht   = (_Float16*)(ws + HT_OFF);
    float*    s_ar = (float*)(ws + S_OFF);
    float*    t_ar = (float*)(ws + T_OFF);
    float*    pacc = (float*)(ws + PACC_OFF);
    float*    pst  = (float*)(ws + PST_OFF);

    k_wtrans<<<(IN_FEAT * OUT_FEAT) / 256, 256, 0, stream>>>(W, Wt);
    k_hmm<<<64, 256, 0, stream>>>(X, Wt, a_self, a_nei, Ht, s_ar, t_ar);
    // 512 row-blocks x 4 column groups = 2048 waves; 128 KB LDS ring / block
    k_attn<<<256, 256, 131072, stream>>>(adj, Mm, Ht, s_ar, t_ar, pacc, pst);
    k_merge<<<(N_NODES * OUT_FEAT) / 256, 256, 0, stream>>>(pacc, pst, out);
}
